// GPT_64201171141386
// MI455X (gfx1250) — matrix operand order repaired and checked by reading, not compiled, not run on hardware
//
#include <hip/hip_runtime.h>
#include <hip/hip_bf16.h>
#include <math.h>

// ---------------------------------------------------------------------------
// GPT-2-small forward for MI455X (gfx1250), wave32 + WMMA f16->f32.
// GEMM tiles staged into LDS by the Tensor Data Mover (tensor_load_to_lds,
// TENSORcnt, hardware row-padding), double-buffered so DMA overlaps WMMA.
// Attention is flash-style per-wave (16 query rows / wave), V stored [B,H,D,S].
// ---------------------------------------------------------------------------

typedef _Float16 half_t;
typedef __attribute__((ext_vector_type(16))) _Float16 v16h;
typedef __attribute__((ext_vector_type(8)))  _Float16 v8h;
typedef __attribute__((ext_vector_type(8)))  float    v8f;
typedef __attribute__((ext_vector_type(4)))  unsigned int v4u;
typedef __attribute__((ext_vector_type(8)))  int      v8i;
typedef __attribute__((ext_vector_type(4)))  int      v4i;

#define E_   768
#define FF_  4608
#define V_   32000
#define L_   4
#define S_   1024
#define B_   2
#define H_   12
#define D_   64
#define M_   2048   // B_*S_

__device__ __forceinline__ v8f wmma16(v16h a, v16h b, v8f c) {
  return __builtin_amdgcn_wmma_f32_16x16x32_f16(
      /*neg_a=*/false, a, /*neg_b=*/false, b,
      /*c_mod=*/(short)0, c, /*reuse_a=*/false, /*reuse_b=*/false);
}

// Issue one TDM tile load: 128 rows x 32 f16 from global (row stride K elems)
// into LDS at lds_addr, with HW padding of 4 DWORDs after every 16 DWORDs
// (=> 40-half padded rows). Descriptor per CDNA5 ISA 8.3/8.4.
// Toolchain exposes the 6-arg builtin: (g0, g1, g2, g3, g_extra, cpol).
__device__ __forceinline__ void tdm_load_tile(const half_t* gbase,
                                              unsigned lds_addr, int K) {
  unsigned long long ga = (unsigned long long)(size_t)gbase;
  v4u g0;
  g0[0] = 1u;                                        // count=1, user mode
  g0[1] = lds_addr;                                  // LDS byte address
  g0[2] = (unsigned)(ga & 0xFFFFFFFFull);            // global_addr[31:0]
  g0[3] = (unsigned)((ga >> 32) & 0x1FFFFFFull)      // global_addr[56:32]
          | 0x80000000u;                             // type=2 ("image")
  v8i g1;
  g1[0] = (1 << 16)     // data_size = 1 (2 bytes)
        | (1 << 20)     // pad_enable
        | (3 << 22)     // pad_interval: 3 -> every 16 DWORDs (64B row)
        | (3 << 25);    // pad_amount:   3 -> 4 DWORDs (8 halfs)
  g1[1] = (K & 0xFFFF) << 16;                        // tensor_dim0[15:0]
  g1[2] = (int)((unsigned)K >> 16);                  // dim1 lo16(=0)<<16 | dim0 hi16
  g1[3] = (32 << 16) | 0x10;                         // tile_dim0=32 | dim1 hi16 (dim1=1M)
  g1[4] = 128;                                       // tile_dim2=0 | tile_dim1=128
  g1[5] = K;                                         // tensor_dim0_stride[31:0]
  g1[6] = 0;                                         // stride1 lo16<<16 | stride0 hi16
  g1[7] = 0;
  v4i z4 = {};
  v8i z8 = {};
  __builtin_amdgcn_tensor_load_to_lds(g0, g1, z4, z4, z8, 0);
}

// ------------------------- weight transpose fp32 -> f16 --------------------
// in: [K][N] fp32 (row-major), out: [N][K] f16. blockIdx.z = layer.
__global__ __launch_bounds__(256) void transpose_to_f16(
    const float* __restrict__ in, half_t* __restrict__ outT, int K, int N) {
  __shared__ float tile[32][33];
  const float* inz  = in  + (size_t)blockIdx.z * K * N;
  half_t*      outz = outT + (size_t)blockIdx.z * K * N;
  const int n0 = blockIdx.x * 32, k0 = blockIdx.y * 32;
  const int tx = threadIdx.x & 31, ty = threadIdx.x >> 5;  // 32 x 8
#pragma unroll
  for (int i = 0; i < 4; ++i)
    tile[ty + i * 8][tx] = inz[(size_t)(k0 + ty + i * 8) * N + n0 + tx];
  __syncthreads();
#pragma unroll
  for (int i = 0; i < 4; ++i)
    outz[(size_t)(n0 + ty + i * 8) * K + k0 + tx] = (half_t)tile[tx][ty + i * 8];
}

// ------------------------- embedding ---------------------------------------
__global__ __launch_bounds__(256) void embed_kernel(
    const int* __restrict__ tokens, const float* __restrict__ tok_emb,
    const float* __restrict__ pos_emb, float* __restrict__ x) {
  const int row = blockIdx.x;          // 0..2047 = b*S + s
  const int s   = row & (S_ - 1);
  const int tok = tokens[row];
  for (int i = threadIdx.x; i < E_; i += 256)
    x[(size_t)row * E_ + i] = tok_emb[(size_t)tok * E_ + i] + pos_emb[(size_t)s * E_ + i];
}

// ------------------------- layernorm (fp32 in, f16 out) --------------------
__global__ __launch_bounds__(256) void layernorm_kernel(
    const float* __restrict__ x, const float* __restrict__ g,
    const float* __restrict__ bta, half_t* __restrict__ out) {
  __shared__ float rs[8], rs2[8];
  const int row = blockIdx.x;
  const float* xr = x + (size_t)row * E_;
  float s = 0.f, s2 = 0.f;
  for (int i = threadIdx.x; i < E_; i += 256) { float v0 = xr[i]; s += v0; s2 += v0 * v0; }
#pragma unroll
  for (int off = 16; off > 0; off >>= 1) { s += __shfl_xor(s, off); s2 += __shfl_xor(s2, off); }
  if ((threadIdx.x & 31) == 0) { rs[threadIdx.x >> 5] = s; rs2[threadIdx.x >> 5] = s2; }
  __syncthreads();
  float ts = 0.f, ts2 = 0.f;
#pragma unroll
  for (int i2 = 0; i2 < 8; ++i2) { ts += rs[i2]; ts2 += rs2[i2]; }
  const float mean = ts * (1.0f / E_);
  const float var  = ts2 * (1.0f / E_) - mean * mean;
  const float inv  = rsqrtf(var + 1e-5f);
  for (int i = threadIdx.x; i < E_; i += 256)
    out[(size_t)row * E_ + i] = (half_t)(((xr[i] - mean) * inv) * g[i] + bta[i]);
}

// ------------------------- WMMA GEMM (TDM-staged, double-buffered) ---------
// C[M,N] = act(((A[M,K] @ B[K,N]) + bias) * scale), B given transposed: BT[N][K].
// Block tile 128x128, BK=32, 256 threads = 8 waves (2 M x 4 N), wave = 64x32.
// Wave 0 drives the Tensor Data Mover; buffers ping-pong so DMA overlaps WMMA.
// STORE: 0=f16 row-major(ldo)  1=QK head-split f16  2=V transposed f16
//        3=fp32 += (residual)  4=fp32 store
template <int STORE, bool ACT_GELU>
__global__ __launch_bounds__(256) void gemm_wmma_kernel(
    const half_t* __restrict__ A, const half_t* __restrict__ BT,
    const float* __restrict__ bias, void* __restrict__ outp,
    int K, int ldo, float scale) {
  __shared__ __align__(16) half_t AsBuf[2][128 * 40];  // 40-half padded rows
  __shared__ __align__(16) half_t BsBuf[2][128 * 40];
  const int t    = threadIdx.x;
  const int m0   = blockIdx.x * 128;
  const int n0   = blockIdx.y * 128;
  const int lane = t & 31;
  const int w    = t >> 5;
  const int wm   = (w >> 2) * 64;   // wave M offset in block
  const int wn   = (w & 3) * 32;    // wave N offset in block
  const int r16  = lane & 15;
  const int hi   = lane >> 4;       // 0/1 half-wave
  const int koffA = hi * 8;         // A frag: halfs [koffA,+8) and [16+koffA,+8)
  const int koffB = hi * 8;         // B frag: halfs [koffB,+8) and [16+koffB,+8)

  const unsigned ldsA[2] = {(unsigned)(size_t)(void*)&AsBuf[0][0],
                            (unsigned)(size_t)(void*)&AsBuf[1][0]};
  const unsigned ldsB[2] = {(unsigned)(size_t)(void*)&BsBuf[0][0],
                            (unsigned)(size_t)(void*)&BsBuf[1][0]};
  const half_t* Ag = A  + (size_t)m0 * K;
  const half_t* Bg = BT + (size_t)n0 * K;

  v8f acc[4][2] = {};

  if (w == 0) {                      // preload tile 0 via TDM
    tdm_load_tile(Ag, ldsA[0], K);
    tdm_load_tile(Bg, ldsB[0], K);
    __builtin_amdgcn_s_wait_tensorcnt((short)0);
  }
  __syncthreads();

  const int nk = K >> 5;
  for (int kt = 0; kt < nk; ++kt) {
    const int cur = kt & 1;
    if (w == 0 && kt + 1 < nk) {     // DMA next tile while computing this one
      tdm_load_tile(Ag + (kt + 1) * 32, ldsA[cur ^ 1], K);
      tdm_load_tile(Bg + (kt + 1) * 32, ldsB[cur ^ 1], K);
    }
    const half_t* As = AsBuf[cur];
    const half_t* Bs = BsBuf[cur];
    v16h a[4], b[2];
#pragma unroll
    for (int i = 0; i < 4; ++i) {
      union { v16h v; v8h h[2]; } u;
      const half_t* p = &As[(wm + i * 16 + r16) * 40];
      u.h[0] = *(const v8h*)&p[koffA];
      u.h[1] = *(const v8h*)&p[16 + koffA];
      a[i] = u.v;
    }
#pragma unroll
    for (int j = 0; j < 2; ++j) {
      union { v16h v; v8h h[2]; } u;
      const half_t* p = &Bs[(wn + j * 16 + r16) * 40];
      u.h[0] = *(const v8h*)&p[koffB];
      u.h[1] = *(const v8h*)&p[16 + koffB];
      b[j] = u.v;
    }
#pragma unroll
    for (int i = 0; i < 4; ++i)
#pragma unroll
      for (int j = 0; j < 2; ++j)
        acc[i][j] = wmma16(a[i], b[j], acc[i][j]);
    if (w == 0) __builtin_amdgcn_s_wait_tensorcnt((short)0);
    __syncthreads();                 // publish next buffer / retire this one
  }

  // Epilogue. C layout: M = r + 8*hi, N = lane&15 per 16x16 tile.
#pragma unroll
  for (int i = 0; i < 4; ++i) {
#pragma unroll
    for (int j = 0; j < 2; ++j) {
      const int n  = n0 + wn + j * 16 + r16;
      const float bv = bias[n];
      const int mbase = m0 + wm + i * 16 + hi * 8;
#pragma unroll
      for (int r = 0; r < 8; ++r) {
        const int m = mbase + r;
        float v = (acc[i][j][r] + bv) * scale;
        if constexpr (ACT_GELU)
          v = 0.5f * v * (1.0f + erff(v * 0.70710678118f));
        if constexpr (STORE == 0) {
          ((half_t*)outp)[(size_t)m * ldo + n] = (half_t)v;
        } else if constexpr (STORE == 1) {  // q/k: [B,H,S,D]
          int b_ = m >> 10, s_ = m & (S_ - 1), h_ = n >> 6, d_ = n & 63;
          ((half_t*)outp)[((((size_t)b_ * H_ + h_) * S_ + s_) << 6) + d_] = (half_t)v;
        } else if constexpr (STORE == 2) {  // v: [B,H,D,S]
          int b_ = m >> 10, s_ = m & (S_ - 1), h_ = n >> 6, d_ = n & 63;
          ((half_t*)outp)[((((size_t)b_ * H_ + h_) * D_ + d_) << 10) + s_] = (half_t)v;
        } else if constexpr (STORE == 3) {  // residual add fp32
          float* p = (float*)outp + (size_t)m * ldo + n;
          *p += v;
        } else {                             // fp32 store (logits)
          ((float*)outp)[(size_t)m * ldo + n] = v;
        }
      }
    }
  }
}

// ------------------------- flash attention ---------------------------------
// grid = (S/128, B*H). Each wave owns 16 query rows, iterates 64-key blocks.
// q pre-scaled by D^-0.5 at store time. v is [B,H,D,S]. Output -> o[M][E].
__global__ __launch_bounds__(256) void attn_kernel(
    const half_t* __restrict__ q, const half_t* __restrict__ k,
    const half_t* __restrict__ v, half_t* __restrict__ o) {
  __shared__ float sS[8][16 * 68];                 // raw scores (padded rows)
  __shared__ float sA[8][16];                      // per-row alpha / 1/l
  __shared__ __align__(16) half_t sP[8][16 * 72];  // probs f16 (padded rows)
  const int t = threadIdx.x, lane = t & 31, w = t >> 5;
  const int bh = blockIdx.y;
  const int b_ = bh / H_;
  const int q0 = blockIdx.x * 128 + w * 16;
  const int r16 = lane & 15, hi = lane >> 4;
  const int koffA = hi * 8, koffB = hi * 16;
  const half_t* qp  = q + ((size_t)bh * S_ + q0) * D_;
  const half_t* kp0 = k + (size_t)bh * S_ * D_;
  const half_t* vp  = v + (size_t)bh * D_ * S_;

  v16h aq[2];
#pragma unroll
  for (int kf = 0; kf < 2; ++kf) {
    union { v16h v; v8h h[2]; } u;
    const half_t* p = qp + r16 * D_ + kf * 32;
    u.h[0] = *(const v8h*)&p[koffA];
    u.h[1] = *(const v8h*)&p[16 + koffA];
    aq[kf] = u.v;
  }

  v8f accO[4] = {};
  const int row = lane >> 1, hc = lane & 1;  // softmax mapping: half-rows
  float m_run = -1e30f, l_run = 0.0f;
  float*  sc = sS[w];
  half_t* sp = sP[w];
  const int kbmax = (q0 + 15) >> 6;

  for (int kb = 0; kb <= kbmax; ++kb) {
    if (kb < kbmax) {  // prefetch next K/V block (global_prefetch_b8)
      __builtin_prefetch(kp0 + (size_t)(kb + 1) * 64 * D_ + lane * 64, 0, 3);
      __builtin_prefetch(vp + (size_t)(lane & 15) * S_ + (kb + 1) * 64, 0, 3);
    }
    v8f accS[4] = {};
    const half_t* kp = kp0 + (size_t)kb * 64 * D_;
#pragma unroll
    for (int jn = 0; jn < 4; ++jn) {
#pragma unroll
      for (int kf = 0; kf < 2; ++kf) {
        union { v16h v; v8h h[2]; } u;
        const half_t* p = kp + (jn * 16 + r16) * D_ + kf * 32;
        u.h[0] = *(const v8h*)&p[koffA];
        u.h[1] = *(const v8h*)&p[16 + koffA];
        accS[jn] = wmma16(aq[kf], u.v, accS[jn]);
      }
    }
    // hoist V fragment loads: latency hidden under softmax VALU work
    v16h vf[4][2];
#pragma unroll
    for (int jd = 0; jd < 4; ++jd)
#pragma unroll
      for (int kf = 0; kf < 2; ++kf) {
        union { v16h v; v8h h[2]; } u;
        const half_t* p = vp + (size_t)(jd * 16 + r16) * S_ + kb * 64 + kf * 32;
        u.h[0] = *(const v8h*)&p[koffA];
        u.h[1] = *(const v8h*)&p[16 + koffA];
        vf[jd][kf] = u.v;
      }
    // causal mask + stage scores to LDS
#pragma unroll
    for (int jn = 0; jn < 4; ++jn) {
      const int kcol = jn * 16 + r16;
      const int kg = kb * 64 + kcol;
#pragma unroll
      for (int r = 0; r < 8; ++r) {
        const int qrow = r + hi * 8;
        const float sv = (kg <= q0 + qrow) ? accS[jn][r] : -1e30f;
        sc[qrow * 68 + kcol] = sv;
      }
    }
    asm volatile("s_wait_dscnt 0" ::: "memory");
    // online softmax: lane pair handles one row (32 cols each)
    const float* srow = sc + row * 68 + hc * 32;
    float lm = -1e30f;
#pragma unroll
    for (int c = 0; c < 32; ++c) lm = fmaxf(lm, srow[c]);
    lm = fmaxf(lm, __shfl_xor(lm, 1));
    const float mnew  = fmaxf(m_run, lm);
    const float alpha = __expf(m_run - mnew);
    float ls = 0.0f;
    half_t* prow = sp + row * 72 + hc * 32;
#pragma unroll
    for (int c = 0; c < 32; ++c) {
      const float pe = __expf(srow[c] - mnew);
      ls += pe;
      prow[c] = (half_t)pe;
    }
    ls += __shfl_xor(ls, 1);
    l_run = l_run * alpha + ls;
    m_run = mnew;
    if (hc == 0) sA[w][row] = alpha;
    asm volatile("s_wait_dscnt 0" ::: "memory");
    float alr[8];
#pragma unroll
    for (int r = 0; r < 8; ++r) alr[r] = sA[w][r + hi * 8];
#pragma unroll
    for (int jd = 0; jd < 4; ++jd)
#pragma unroll
      for (int r = 0; r < 8; ++r) accO[jd][r] *= alr[r];
    // P (A-frag) from LDS, V fragments already in registers
    v16h ap[2];
#pragma unroll
    for (int kf = 0; kf < 2; ++kf) {
      union { v16h v; v8h h[2]; } u;
      const half_t* p = sp + r16 * 72 + kf * 32;
      u.h[0] = *(const v8h*)&p[koffA];
      u.h[1] = *(const v8h*)&p[16 + koffA];
      ap[kf] = u.v;
    }
#pragma unroll
    for (int jd = 0; jd < 4; ++jd)
#pragma unroll
      for (int kf = 0; kf < 2; ++kf)
        accO[jd] = wmma16(ap[kf], vf[jd][kf], accO[jd]);
  }
  if (hc == 0) sA[w][row] = 1.0f / l_run;
  asm volatile("s_wait_dscnt 0" ::: "memory");
  float lr[8];
#pragma unroll
  for (int r = 0; r < 8; ++r) lr[r] = sA[w][r + hi * 8];
  const int hcol0 = (bh % H_) * D_;
#pragma unroll
  for (int jd = 0; jd < 4; ++jd)
#pragma unroll
    for (int r = 0; r < 8; ++r) {
      const int m   = b_ * S_ + q0 + r + hi * 8;
      const int col = hcol0 + jd * 16 + r16;
      o[(size_t)m * E_ + col] = (half_t)(accO[jd][r] * lr[r]);
    }
}

// ------------------------- workspace layout --------------------------------
constexpr size_t OFF_X   = 0;                                // fp32 [M][E]
constexpr size_t OFF_H   = OFF_X   + (size_t)M_ * E_ * 4;    // f16  [M][E]
constexpr size_t OFF_Q   = OFF_H   + (size_t)M_ * E_ * 2;    // f16  [B,H,S,D]
constexpr size_t OFF_K   = OFF_Q   + (size_t)M_ * E_ * 2;
constexpr size_t OFF_V   = OFF_K   + (size_t)M_ * E_ * 2;    // f16  [B,H,D,S]
constexpr size_t OFF_O   = OFF_V   + (size_t)M_ * E_ * 2;    // f16  [M][E]
constexpr size_t OFF_FF  = OFF_O   + (size_t)M_ * E_ * 2;    // f16  [M][FF]
constexpr size_t OFF_WQT = OFF_FF  + (size_t)M_ * FF_ * 2;
constexpr size_t OFF_WKT = OFF_WQT + (size_t)L_ * E_ * E_ * 2;
constexpr size_t OFF_WVT = OFF_WKT + (size_t)L_ * E_ * E_ * 2;
constexpr size_t OFF_WOT = OFF_WVT + (size_t)L_ * E_ * E_ * 2;
constexpr size_t OFF_W1T = OFF_WOT + (size_t)L_ * E_ * E_ * 2;
constexpr size_t OFF_W2T = OFF_W1T + (size_t)L_ * E_ * FF_ * 2;
constexpr size_t OFF_WFT = OFF_W2T + (size_t)L_ * E_ * FF_ * 2;

extern "C" void kernel_launch(void* const* d_in, const int* in_sizes, int n_in,
                              void* d_out, int out_size, void* d_ws, size_t ws_size,
                              hipStream_t stream) {
  const int*   tokens  = (const int*)  d_in[0];
  const float* tok_emb = (const float*)d_in[1];
  const float* pos_emb = (const float*)d_in[2];
  const float* Wq = (const float*)d_in[3];
  const float* bq = (const float*)d_in[4];
  const float* Wk = (const float*)d_in[5];
  const float* bk = (const float*)d_in[6];
  const float* Wv = (const float*)d_in[7];
  const float* bv = (const float*)d_in[8];
  const float* Wo = (const float*)d_in[9];
  const float* bo = (const float*)d_in[10];
  const float* ln1_g = (const float*)d_in[11];
  const float* ln1_b = (const float*)d_in[12];
  const float* ln2_g = (const float*)d_in[13];
  const float* ln2_b = (const float*)d_in[14];
  const float* W1 = (const float*)d_in[15];
  const float* b1 = (const float*)d_in[16];
  const float* W2 = (const float*)d_in[17];
  const float* b2 = (const float*)d_in[18];
  const float* lnf_g = (const float*)d_in[19];
  const float* lnf_b = (const float*)d_in[20];
  const float* Wf = (const float*)d_in[21];
  const float* bf = (const float*)d_in[22];

  char* ws = (char*)d_ws;
  float*  xb  = (float*) (ws + OFF_X);
  half_t* hb  = (half_t*)(ws + OFF_H);
  half_t* qb  = (half_t*)(ws + OFF_Q);
  half_t* kbf = (half_t*)(ws + OFF_K);
  half_t* vbf = (half_t*)(ws + OFF_V);
  half_t* ob  = (half_t*)(ws + OFF_O);
  half_t* ffb = (half_t*)(ws + OFF_FF);
  half_t* wqT = (half_t*)(ws + OFF_WQT);
  half_t* wkT = (half_t*)(ws + OFF_WKT);
  half_t* wvT = (half_t*)(ws + OFF_WVT);
  half_t* woT = (half_t*)(ws + OFF_WOT);
  half_t* w1T = (half_t*)(ws + OFF_W1T);
  half_t* w2T = (half_t*)(ws + OFF_W2T);
  half_t* wfT = (half_t*)(ws + OFF_WFT);

  const dim3 tb(256);
  // Weight conversion (deterministic, every call)
  transpose_to_f16<<<dim3(E_ / 32,  E_ / 32,  L_), tb, 0, stream>>>(Wq, wqT, E_, E_);
  transpose_to_f16<<<dim3(E_ / 32,  E_ / 32,  L_), tb, 0, stream>>>(Wk, wkT, E_, E_);
  transpose_to_f16<<<dim3(E_ / 32,  E_ / 32,  L_), tb, 0, stream>>>(Wv, wvT, E_, E_);
  transpose_to_f16<<<dim3(E_ / 32,  E_ / 32,  L_), tb, 0, stream>>>(Wo, woT, E_, E_);
  transpose_to_f16<<<dim3(FF_ / 32, E_ / 32,  L_), tb, 0, stream>>>(W1, w1T, E_, FF_);
  transpose_to_f16<<<dim3(E_ / 32,  FF_ / 32, L_), tb, 0, stream>>>(W2, w2T, FF_, E_);
  transpose_to_f16<<<dim3(V_ / 32,  E_ / 32,  1 ), tb, 0, stream>>>(Wf, wfT, E_, V_);

  embed_kernel<<<M_, tb, 0, stream>>>(tokens, tok_emb, pos_emb, xb);

  for (int l = 0; l < L_; ++l) {
    const size_t wo_ = (size_t)l * E_ * E_;
    const size_t wf1 = (size_t)l * E_ * FF_;
    layernorm_kernel<<<M_, tb, 0, stream>>>(xb, ln1_g + l * E_, ln1_b + l * E_, hb);
    // QKV projections (q pre-scaled by D^-0.5)
    gemm_wmma_kernel<1, false><<<dim3(M_ / 128, E_ / 128), tb, 0, stream>>>(
        hb, wqT + wo_, bq + l * E_, qb, E_, 0, 0.125f);
    gemm_wmma_kernel<1, false><<<dim3(M_ / 128, E_ / 128), tb, 0, stream>>>(
        hb, wkT + wo_, bk + l * E_, kbf, E_, 0, 1.0f);
    gemm_wmma_kernel<2, false><<<dim3(M_ / 128, E_ / 128), tb, 0, stream>>>(
        hb, wvT + wo_, bv + l * E_, vbf, E_, 0, 1.0f);
    attn_kernel<<<dim3(S_ / 128, B_ * H_), tb, 0, stream>>>(qb, kbf, vbf, ob);
    // output projection + residual
    gemm_wmma_kernel<3, false><<<dim3(M_ / 128, E_ / 128), tb, 0, stream>>>(
        ob, woT + wo_, bo + l * E_, xb, E_, E_, 1.0f);
    layernorm_kernel<<<M_, tb, 0, stream>>>(xb, ln2_g + l * E_, ln2_b + l * E_, hb);
    // MLP
    gemm_wmma_kernel<0, true><<<dim3(M_ / 128, FF_ / 128), tb, 0, stream>>>(
        hb, w1T + wf1, b1 + l * FF_, ffb, E_, FF_, 1.0f);
    gemm_wmma_kernel<3, false><<<dim3(M_ / 128, E_ / 128), tb, 0, stream>>>(
        ffb, w2T + wf1, b2 + l * E_, xb, FF_, E_, 1.0f);
  }

  layernorm_kernel<<<M_, tb, 0, stream>>>(xb, lnf_g, lnf_b, hb);
  gemm_wmma_kernel<4, false><<<dim3(M_ / 128, V_ / 128), tb, 0, stream>>>(
      hb, wfT, bf, d_out, E_, V_, 1.0f);
}
